// LinkPredict_56599079026724
// MI455X (gfx1250) — matrix operand order repaired and checked by reading, not compiled, not run on hardware
//
#include <hip/hip_runtime.h>

// ---------------- CDNA5 WMMA types / helpers ----------------
typedef __attribute__((ext_vector_type(16))) __bf16 bf16x16;
typedef __attribute__((ext_vector_type(8)))  float  f32x8;

union FragU { bf16x16 v; uint4 u[2]; };

__device__ inline __bf16 f2bf(float f) {
    unsigned u = __builtin_bit_cast(unsigned, f);
    unsigned r = (u + 0x7FFFu + ((u >> 16) & 1u)) >> 16;
    unsigned short s = (unsigned short)r;
    return __builtin_bit_cast(__bf16, s);
}

// A fragment (16x32 bf16, M x K): lane m = lane&15, hi = lane>>4.
// elems 0..7  -> k = hi*8 + j        (contiguous run)
// elems 8..15 -> k = 16 + hi*8 + j   (contiguous run)
__device__ inline bf16x16 load_frag_a(const __bf16* base, int hi) {
    FragU f;
    f.u[0] = *(const uint4*)(base + hi * 8);
    f.u[1] = *(const uint4*)(base + 16 + hi * 8);
    return f.v;
}
// B fragment (32x16 bf16, K x N): lane n = lane&15, same k order as A:
// elems 0..7 -> k = hi*8 + j, elems 8..15 -> k = 16 + hi*8 + j. base points at row n.
__device__ inline bf16x16 load_frag_b(const __bf16* base, int hi) {
    FragU f;
    const uint4* p = (const uint4*)base;
    f.u[0] = p[hi];
    f.u[1] = p[2 + hi];
    return f.v;
}

__device__ inline f32x8 wmma_bf16(bf16x16 a, bf16x16 b, f32x8 c) {
    return __builtin_amdgcn_wmma_f32_16x16x32_bf16(false, a, false, b, (short)0, c, false, false);
}

// ---------------- constants ----------------
#define BB 512
#define TT 10
#define NN 50
#define HH 200
#define EE 10000

// ---------------- generic GEMM: C[M x Ncols] = A_bf16[M x lda] @ W[Ncols x K]^T (+bias) ----------
// grid.x = M/64, grid.y = ceil(Ncols/16); block = 128 (4 waves x 16 rows).
// A must be zero-padded to lda (multiple of 32). W is f32 row-major with row stride wld.
__global__ void gemm64x16(const __bf16* __restrict__ A, int lda,
                          const float* __restrict__ W, int wld, int K, int Ncols,
                          const float* __restrict__ bias,
                          float* __restrict__ C, int ldc)
{
    __shared__ __bf16 Bs[16 * 608];
    const int tid = threadIdx.x;
    const int c0  = blockIdx.y * 16;
    const int tot = 16 * lda;
    for (int i = tid; i < tot; i += 128) {
        int n = i / lda, k = i - n * lda;
        int g = c0 + n;
        float v = (g < Ncols && k < K) ? W[(size_t)g * wld + k] : 0.f;
        Bs[i] = f2bf(v);
    }
    __syncthreads();

    const int wid = tid >> 5, lane = tid & 31;
    const int mrow = lane & 15, hi = lane >> 4;
    const int row0 = blockIdx.x * 64 + wid * 16;
    const __bf16* arow = A + (size_t)(row0 + mrow) * lda;
    const __bf16* brow = Bs + mrow * lda;

    f32x8 acc = {};
    const int ksteps = lda >> 5;
    for (int ks = 0; ks < ksteps; ++ks) {
        bf16x16 af = load_frag_a(arow + ks * 32, hi);
        bf16x16 bf = load_frag_b(brow + ks * 32, hi);
        acc = wmma_bf16(af, bf, acc);
    }
    int g = c0 + mrow;
    if (g < Ncols) {
        float bv = bias ? bias[g] : 0.f;
        for (int e = 0; e < 8; ++e) {
            int m = e + 8 * hi;
            C[(size_t)(row0 + m) * ldc + g] = acc[e] + bv;
        }
    }
}

// ---------------- attention logits: fused gather + WMMA + tanh.v + reduce ----------------
// rows = b*T*N flattened (256000). grid.x = 4000 (64 rows/block), block = 128.
__global__ void attn_logits(const int* __restrict__ neigh, const int* __restrict__ nlen,
                            const float* __restrict__ ent, const float* __restrict__ Wa,
                            const float* __restrict__ vvec, const float* __restrict__ biasws,
                            float* __restrict__ logits)
{
    __shared__ __bf16 As[64 * 224];
    __shared__ __bf16 Ws_[16 * 224];
    __shared__ int nids[64];
    __shared__ int bidx[64];

    const int tid = threadIdx.x;
    const int rbase = blockIdx.x * 64;
    if (tid < 64) {
        int r = rbase + tid;
        nids[tid] = neigh[r];
        bidx[tid] = r / (TT * NN);
    }
    __syncthreads();
    for (int i = tid; i < 64 * 224; i += 128) {
        int lr = i / 224, k = i - lr * 224;
        As[i] = f2bf(k < HH ? ent[(size_t)nids[lr] * HH + k] : 0.f);
    }

    const int wid = tid >> 5, lane = tid & 31;
    const int mrow = lane & 15, hi = lane >> 4;
    const int lr0 = wid * 16;
    float lacc[8] = {0.f, 0.f, 0.f, 0.f, 0.f, 0.f, 0.f, 0.f};

    for (int gt = 0; gt < 13; ++gt) {
        __syncthreads();
        for (int i = tid; i < 16 * 224; i += 128) {
            int n = i / 224, k = i - n * 224;
            int g = gt * 16 + n;
            Ws_[i] = f2bf((g < HH && k < HH) ? Wa[(size_t)g * 600 + k] : 0.f);
        }
        __syncthreads();

        f32x8 acc = {};
        const __bf16* arow = As + (lr0 + mrow) * 224;
        const __bf16* brow = Ws_ + mrow * 224;
        for (int ks = 0; ks < 7; ++ks) {
            bf16x16 af = load_frag_a(arow + ks * 32, hi);
            bf16x16 bf = load_frag_b(brow + ks * 32, hi);
            acc = wmma_bf16(af, bf, acc);
        }
        int g = gt * 16 + mrow;
        float gv = (g < HH) ? vvec[g] : 0.f;
        for (int e = 0; e < 8; ++e) {
            int m = e + 8 * hi;
            float bi = (g < HH) ? biasws[(size_t)bidx[lr0 + m] * HH + g] : 0.f;
            lacc[e] += gv * tanhf(acc[e] + bi);
        }
    }
    // reduce each row's partial across the 16 lanes of each half-wave
    for (int e = 0; e < 8; ++e) {
        float s = lacc[e];
        s += __shfl_xor(s, 1, 32);
        s += __shfl_xor(s, 2, 32);
        s += __shfl_xor(s, 4, 32);
        s += __shfl_xor(s, 8, 32);
        lacc[e] = s;
    }
    if (mrow == 0) {
        for (int e = 0; e < 8; ++e) {
            int m = e + 8 * hi;
            int r = rbase + lr0 + m;
            int n = r % NN;
            int bt = r / NN;
            float lg = lacc[e];
            if (n >= nlen[bt]) lg = -1e30f;
            logits[r] = lg;
        }
    }
}

// ---------------- softmax over neighbors + weighted step + seq assembly (bf16, padded 608) -------
__global__ void softmax_step(const int* __restrict__ neigh, const int* __restrict__ trip,
                             const int* __restrict__ histlen,
                             const float* __restrict__ ent, const float* __restrict__ rel,
                             const float* __restrict__ logits,
                             __bf16* __restrict__ seq, int ent_col)
{
    __shared__ float wgt[NN];
    __shared__ int   nid[NN];
    __shared__ float lg[NN];
    __shared__ float mxs, invs;

    const int bt = blockIdx.x;
    const int b = bt / TT, t = bt - b * TT;
    const int tid = threadIdx.x;
    if (tid < NN) {
        nid[tid] = neigh[bt * NN + tid];
        lg[tid]  = logits[bt * NN + tid];
    }
    __syncthreads();
    if (tid == 0) {
        float m = lg[0];
        for (int i = 1; i < NN; ++i) m = fmaxf(m, lg[i]);
        mxs = m;
    }
    __syncthreads();
    if (tid < NN) wgt[tid] = expf(lg[tid] - mxs);
    __syncthreads();
    if (tid == 0) {
        float s = 0.f;
        for (int i = 0; i < NN; ++i) s += wgt[i];
        invs = 1.f / s;
    }
    __syncthreads();

    const float sm = (t < histlen[b]) ? 1.f : 0.f;
    __bf16* row = seq + (size_t)bt * 608;
    const int sv = trip[b * 3 + ent_col];
    const int rv = trip[b * 3 + 1];
    if (tid < HH) {
        float a = 0.f;
        for (int n = 0; n < NN; ++n)
            a += wgt[n] * invs * ent[(size_t)nid[n] * HH + tid];
        row[tid]          = f2bf(a * sm);
        row[HH + tid]     = f2bf(ent[(size_t)sv * HH + tid] * sm);
        row[2 * HH + tid] = f2bf(rel[(size_t)rv * HH + tid] * sm);
    } else if (tid < 208) {
        row[600 + (tid - 200)] = f2bf(0.f);
    }
}

// ---------------- misc small kernels ----------------
__global__ void prep_biasA(const int* __restrict__ trip, const float* __restrict__ ent,
                           __bf16* __restrict__ out, int ent_col)
{
    int idx = blockIdx.x * 256 + threadIdx.x;      // 512*416
    if (idx >= BB * 416) return;
    int b = idx / 416, k = idx - b * 416;
    float v;
    if (k < HH)           v = ent[(size_t)trip[b * 3 + ent_col] * HH + k];
    else if (k < 2 * HH)  v = ent[(size_t)trip[b * 3 + 1] * HH + (k - HH)];
    else                  v = 0.f;
    out[idx] = f2bf(v);
}

__global__ void zero_u32(unsigned* __restrict__ p, int n)
{
    int i = blockIdx.x * 256 + threadIdx.x;
    if (i < n) p[i] = 0u;
}

__global__ void gru_update(const float* __restrict__ gi, const float* __restrict__ gh,
                           const int* __restrict__ lens, float* __restrict__ hf,
                           __bf16* __restrict__ hbf, int t)
{
    int idx = blockIdx.x * 256 + threadIdx.x;      // 512*200
    if (idx >= BB * HH) return;
    int b = idx / HH, hc = idx - b * HH;
    const float* gir = gi + (size_t)(b * TT + t) * 608;
    const float* ghr = gh + (size_t)b * 608;
    float ir = gir[hc], iz = gir[HH + hc], inn = gir[2 * HH + hc];
    float hr = ghr[hc], hz = ghr[HH + hc], hn  = ghr[2 * HH + hc];
    float hprev = hf[idx];
    float rg = 1.f / (1.f + expf(-(ir + hr)));
    float zg = 1.f / (1.f + expf(-(iz + hz)));
    float ng = tanhf(inn + rg * hn);
    float hnew = (1.f - zg) * ng + zg * hprev;
    float ho = (t < lens[b]) ? hnew : hprev;
    hf[idx] = ho;
    hbf[(size_t)b * 224 + hc] = f2bf(ho);
}

// stable descending argsort via rank counting (matches jnp.argsort(-len), stable ties)
__global__ void argsort_desc(const int* __restrict__ lens, int* __restrict__ idx_out,
                             float* __restrict__ fout)
{
    __shared__ int L[BB];
    int j = threadIdx.x;
    L[j] = lens[j];
    __syncthreads();
    int lj = L[j];
    int rank = 0;
    for (int i = 0; i < BB; ++i) {
        int li = L[i];
        rank += (li > lj) || (li == lj && i < j);
    }
    idx_out[rank] = j;
    fout[rank] = (float)j;
}

__global__ void concat_x(const int* __restrict__ trip, const int* __restrict__ idxarr,
                         const float* __restrict__ ent, const float* __restrict__ rel,
                         const float* __restrict__ hf, __bf16* __restrict__ out, int ent_col)
{
    int idx = blockIdx.x * 256 + threadIdx.x;      // 512*608
    if (idx >= BB * 608) return;
    int bp = idx / 608, k = idx - bp * 608;
    int src = idxarr[bp];
    float v;
    if (k < HH)           v = ent[(size_t)trip[src * 3 + ent_col] * HH + k];
    else if (k < 2 * HH)  v = hf[src * HH + (k - HH)];
    else if (k < 3 * HH)  v = rel[(size_t)trip[src * 3 + 1] * HH + (k - 2 * HH)];
    else                  v = 0.f;
    out[idx] = f2bf(v);
}

__global__ void ce_rows(const float* __restrict__ pred, const int* __restrict__ trip,
                        const int* __restrict__ idxarr, int label_col,
                        float* __restrict__ partial)
{
    __shared__ float red[256];
    const int row = blockIdx.x;
    const int tid = threadIdx.x;
    const float* p = pred + (size_t)row * EE;
    float m = -3.4e38f;
    for (int i = tid; i < EE; i += 256) m = fmaxf(m, p[i]);
    red[tid] = m; __syncthreads();
    for (int s = 128; s > 0; s >>= 1) {
        if (tid < s) red[tid] = fmaxf(red[tid], red[tid + s]);
        __syncthreads();
    }
    float mx = red[0]; __syncthreads();
    float sum = 0.f;
    for (int i = tid; i < EE; i += 256) sum += expf(p[i] - mx);
    red[tid] = sum; __syncthreads();
    for (int s = 128; s > 0; s >>= 1) {
        if (tid < s) red[tid] += red[tid + s];
        __syncthreads();
    }
    if (tid == 0) {
        int lab = trip[idxarr[row] * 3 + label_col];
        float lp = p[lab] - mx - logf(red[0]);
        partial[row] = -lp;
    }
}

__global__ void loss_reduce(const float* __restrict__ partial, float* __restrict__ out)
{
    if (threadIdx.x == 0) {
        float a = 0.f, b = 0.f;
        for (int i = 0; i < BB; ++i) a += partial[i];
        for (int i = 0; i < BB; ++i) b += partial[BB + i];
        out[0] = a / (float)BB + b / (float)BB;
    }
}

// ---------------- host launcher ----------------
extern "C" void kernel_launch(void* const* d_in, const int* in_sizes, int n_in,
                              void* d_out, int out_size, void* d_ws, size_t ws_size,
                              hipStream_t stream)
{
    (void)in_sizes; (void)n_in; (void)out_size; (void)ws_size;
    const int*   trip    = (const int*)d_in[0];
    const int*   s_neigh = (const int*)d_in[1];
    const int*   s_nlen  = (const int*)d_in[2];
    const int*   s_hlen  = (const int*)d_in[3];
    const int*   o_neigh = (const int*)d_in[4];
    const int*   o_nlen  = (const int*)d_in[5];
    const int*   o_hlen  = (const int*)d_in[6];
    const float* ent     = (const float*)d_in[7];
    const float* rel     = (const float*)d_in[8];
    const float* a_s_w   = (const float*)d_in[9];
    const float* a_s_b   = (const float*)d_in[10];
    const float* v_s     = (const float*)d_in[11];
    const float* a_o_w   = (const float*)d_in[12];
    const float* a_o_b   = (const float*)d_in[13];
    const float* v_o     = (const float*)d_in[14];
    const float* sub_wih = (const float*)d_in[15];
    const float* sub_whh = (const float*)d_in[16];
    const float* sub_bih = (const float*)d_in[17];
    const float* sub_bhh = (const float*)d_in[18];
    const float* ob_wih  = (const float*)d_in[19];
    const float* ob_whh  = (const float*)d_in[20];
    const float* ob_bih  = (const float*)d_in[21];
    const float* ob_bhh  = (const float*)d_in[22];
    const float* lin_s_w = (const float*)d_in[23];
    const float* lin_s_b = (const float*)d_in[24];
    const float* lin_o_w = (const float*)d_in[25];
    const float* lin_o_b = (const float*)d_in[26];

    // workspace bump allocator (256B aligned)
    char* wp = (char*)d_ws;
    auto alloc = [&](size_t bytes) -> char* {
        char* p = wp;
        wp += (bytes + 255) & ~(size_t)255;
        return p;
    };
    float*  biasS   = (float*) alloc((size_t)BB * HH * 4);
    float*  biasO   = (float*) alloc((size_t)BB * HH * 4);
    __bf16* AbiasS  = (__bf16*)alloc((size_t)BB * 416 * 2);
    __bf16* AbiasO  = (__bf16*)alloc((size_t)BB * 416 * 2);
    float*  logitsS = (float*) alloc((size_t)BB * TT * NN * 4);
    float*  logitsO = (float*) alloc((size_t)BB * TT * NN * 4);
    __bf16* seqS    = (__bf16*)alloc((size_t)BB * TT * 608 * 2);
    __bf16* seqO    = (__bf16*)alloc((size_t)BB * TT * 608 * 2);
    float*  giS     = (float*) alloc((size_t)BB * TT * 608 * 4);
    float*  giO     = (float*) alloc((size_t)BB * TT * 608 * 4);
    float*  ghS     = (float*) alloc((size_t)BB * 608 * 4);
    float*  ghO     = (float*) alloc((size_t)BB * 608 * 4);
    float*  hfS     = (float*) alloc((size_t)BB * HH * 4);   // contiguous zero region start
    float*  hfO     = (float*) alloc((size_t)BB * HH * 4);
    __bf16* hbfS    = (__bf16*)alloc((size_t)BB * 224 * 2);
    __bf16* hbfO    = (__bf16*)alloc((size_t)BB * 224 * 2);  // contiguous zero region end
    __bf16* XcatS   = (__bf16*)alloc((size_t)BB * 608 * 2);
    __bf16* XcatO   = (__bf16*)alloc((size_t)BB * 608 * 2);
    int*    sidx    = (int*)   alloc((size_t)BB * 4);
    int*    oidx    = (int*)   alloc((size_t)BB * 4);
    float*  partial = (float*) alloc((size_t)2 * BB * 4);

    float* out     = (float*)d_out;
    float* losso   = out;
    float* subpred = out + 1;                           // [512 x 10000]
    float* obpred  = out + 1 + (size_t)BB * EE;         // [512 x 10000]
    float* oidxf   = out + 1 + (size_t)2 * BB * EE;     // [512]
    float* sidxf   = oidxf + BB;                        // [512]

    // 1) attention bias: [se|re] @ Wa[:,200:600]^T + ba   (K=400 pad 416)
    prep_biasA<<<(BB * 416 + 255) / 256, 256, 0, stream>>>(trip, ent, AbiasS, 0);
    prep_biasA<<<(BB * 416 + 255) / 256, 256, 0, stream>>>(trip, ent, AbiasO, 2);
    gemm64x16<<<dim3(BB / 64, 13), 128, 0, stream>>>(AbiasS, 416, a_s_w + 200, 600, 400, HH, a_s_b, biasS, HH);
    gemm64x16<<<dim3(BB / 64, 13), 128, 0, stream>>>(AbiasO, 416, a_o_w + 200, 600, 400, HH, a_o_b, biasO, HH);

    // 2) attention logits (the big einsum): 256000 rows, fused tanh.v epilogue
    attn_logits<<<(BB * TT * NN) / 64, 128, 0, stream>>>(s_neigh, s_nlen, ent, a_s_w, v_s, biasS, logitsS);
    attn_logits<<<(BB * TT * NN) / 64, 128, 0, stream>>>(o_neigh, o_nlen, ent, a_o_w, v_o, biasO, logitsO);

    // 3) softmax + weighted step + seq assembly (bf16, padded to 608)
    softmax_step<<<BB * TT, 256, 0, stream>>>(s_neigh, trip, s_hlen, ent, rel, logitsS, seqS, 0);
    softmax_step<<<BB * TT, 256, 0, stream>>>(o_neigh, trip, o_hlen, ent, rel, logitsO, seqO, 2);

    // 4) GRU input projections for all timesteps at once: gi = seq @ Wih^T + bih
    gemm64x16<<<dim3((BB * TT) / 64, 38), 128, 0, stream>>>(seqS, 608, sub_wih, 600, 600, 600, sub_bih, giS, 608);
    gemm64x16<<<dim3((BB * TT) / 64, 38), 128, 0, stream>>>(seqO, 608, ob_wih, 600, 600, 600, ob_bih, giO, 608);

    // 5) zero GRU state (hfS, hfO, hbfS, hbfO are contiguous in ws)
    {
        size_t zbytes = (size_t)2 * BB * HH * 4 + (size_t)2 * BB * 224 * 2;
        int n32 = (int)(zbytes / 4);
        zero_u32<<<(n32 + 255) / 256, 256, 0, stream>>>((unsigned*)hfS, n32);
    }

    // 6) GRU recurrence (T sequential steps)
    for (int t = 0; t < TT; ++t) {
        gemm64x16<<<dim3(BB / 64, 38), 128, 0, stream>>>(hbfS, 224, sub_whh, 200, 200, 600, sub_bhh, ghS, 608);
        gru_update<<<(BB * HH) / 256, 256, 0, stream>>>(giS, ghS, s_hlen, hfS, hbfS, t);
        gemm64x16<<<dim3(BB / 64, 38), 128, 0, stream>>>(hbfO, 224, ob_whh, 200, 200, 600, ob_bhh, ghO, 608);
        gru_update<<<(BB * HH) / 256, 256, 0, stream>>>(giO, ghO, o_hlen, hfO, hbfO, t);
    }

    // 7) stable descending argsort of history lengths
    argsort_desc<<<1, BB, 0, stream>>>(s_hlen, sidx, sidxf);
    argsort_desc<<<1, BB, 0, stream>>>(o_hlen, oidx, oidxf);

    // 8) build decoder inputs: [ent[s[idx]], h[idx], rel[r[idx]]]
    concat_x<<<(BB * 608) / 256, 256, 0, stream>>>(trip, sidx, ent, rel, hfS, XcatS, 0);
    concat_x<<<(BB * 608) / 256, 256, 0, stream>>>(trip, oidx, ent, rel, hfO, XcatO, 2);

    // 9) decoder GEMMs straight into d_out
    gemm64x16<<<dim3(BB / 64, 625), 128, 0, stream>>>(XcatS, 608, lin_s_w, 600, 600, EE, lin_s_b, obpred, EE);
    gemm64x16<<<dim3(BB / 64, 625), 128, 0, stream>>>(XcatO, 608, lin_o_w, 600, 600, EE, lin_o_b, subpred, EE);

    // 10) cross-entropy (per-row partials, then deterministic serial reduce)
    ce_rows<<<BB, 256, 0, stream>>>(obpred, trip, sidx, 2, partial);          // labels o[s_idx]
    ce_rows<<<BB, 256, 0, stream>>>(subpred, trip, oidx, 0, partial + BB);    // labels s[o_idx]
    loss_reduce<<<1, 32, 0, stream>>>(partial, losso);
}
